// WindowAttention_45887430590845
// MI455X (gfx1250) — matrix operand order repaired and checked by reading, compiled, not run on hardware
//
#include <hip/hip_runtime.h>
#include <hip/hip_bf16.h>

typedef __attribute__((ext_vector_type(16))) _Float16 v16h;
typedef __attribute__((ext_vector_type(8)))  float    v8f;

#define NTOK    64
#define HD      32
#define DIM     512
#define QKV_LD  1536
#define NWIN    49

// ---------------------------------------------------------------------------
// WMMA helpers (CDNA5 v_wmma_f32_16x16x32_f16, wave32)
// ---------------------------------------------------------------------------
__device__ inline v8f wmma16x16x32(v16h a, v16h b, v8f c) {
  return __builtin_amdgcn_wmma_f32_16x16x32_f16(false, a, false, b, (short)0, c,
                                                false, false);
}

// A fragment: 16x32 f16 tile, rows row0..row0+15, cols k0..k0+31, LDS row-major.
// Contiguous k order (same as frag_b/frag_bt): lane lo -> row; slot j -> k = hi*16 + j
// (one contiguous 16-half run per lane -> compiler emits 2x ds_load_b128)
__device__ inline v16h frag_a(const _Float16* p, int row0, int stride, int k0,
                              int lane) {
  const int lo = lane & 15, hi = lane >> 4;
  const _Float16* r = p + (row0 + lo) * stride + k0;
  v16h a;
#pragma unroll
  for (int j = 0; j < 16; ++j) {
    int k = j + hi * 16;
    a[j] = r[k];
  }
  return a;
}

// B fragment: 32x16 f16 tile, B[k][n] stored row-major [k][n] in LDS.
// Contiguous k order (same as frag_a): lane lo -> col n; slot j -> k = hi*16 + j
__device__ inline v16h frag_b(const _Float16* p, int k0, int col0, int stride,
                              int lane) {
  const int lo = lane & 15, hi = lane >> 4;
  v16h b;
#pragma unroll
  for (int j = 0; j < 16; ++j)
    b[j] = p[(k0 + hi * 16 + j) * stride + col0 + lo];
  return b;
}

// B fragment where memory holds the transpose: B[k][n] = p[n][k]
__device__ inline v16h frag_bt(const _Float16* p, int k0, int col0, int stride,
                               int lane) {
  const int lo = lane & 15, hi = lane >> 4;
  v16h b;
#pragma unroll
  for (int j = 0; j < 16; ++j)
    b[j] = p[(col0 + lo) * stride + k0 + hi * 16 + j];
  return b;
}

// ---------------------------------------------------------------------------
// Async DMA: global -> LDS, 16 bytes per lane (GV mode: 64-bit vaddr, no
// saddr).  LDS aperture keeps the LDS byte offset in addr[31:0], so a flat
// pointer truncation yields the VDST operand.  Tracked by ASYNCcnt.
// ---------------------------------------------------------------------------
__device__ inline unsigned lds_off(const void* p) {
  return (unsigned)(unsigned long long)p;
}

__device__ inline void async_copy_b128(unsigned lds_byte_off, const void* g) {
  asm volatile("global_load_async_to_lds_b128 %0, %1, off"
               :: "v"(lds_byte_off), "v"((unsigned long long)g)
               : "memory");
}

__device__ inline void wait_asynccnt0() {
  asm volatile("s_wait_asynccnt 0x0" ::: "memory");
}

// ---------------------------------------------------------------------------
// Continuous relative position bias: MLP(2->384->16) over 15x15 table,
// gathered through rel_index, 16*sigmoid.  One block per head.
// ---------------------------------------------------------------------------
__global__ __launch_bounds__(256) void bias_kernel(
    const float* __restrict__ coords, const float* __restrict__ w1,
    const float* __restrict__ b1, const float* __restrict__ w2,
    const int* __restrict__ rel_index, float* __restrict__ bias_ws) {
  __shared__ float tbl[225];
  const int h = blockIdx.x;
  const int tid = threadIdx.x;
  if (tid < 225) {
    float c0 = coords[tid * 2 + 0], c1 = coords[tid * 2 + 1];
    float acc = 0.f;
    for (int j = 0; j < 384; ++j) {
      float hv = fmaxf(c0 * w1[j * 2 + 0] + c1 * w1[j * 2 + 1] + b1[j], 0.f);
      acc += hv * w2[h * 384 + j];
    }
    tbl[tid] = acc;
  }
  __syncthreads();
  for (int e = tid; e < 4096; e += 256) {
    int idx = rel_index[e];
    bias_ws[h * 4096 + e] = 16.0f / (1.0f + expf(-tbl[idx]));
  }
}

// ---------------------------------------------------------------------------
// GEMM: out[M x Nblk] = A(M x 512) @ W(N x 512)^T + bias, WMMA f16 with f32
// accumulate.  Block tile 128x64, 8 waves (wave w = row strip 16w..16w+15).
// f16 A source uses async-DMA staging; f32 A converts through VGPRs.
// ---------------------------------------------------------------------------
template <typename AT, typename OT>
__global__ __launch_bounds__(256) void gemm_bias_kernel(
    const AT* __restrict__ A, const float* __restrict__ W,
    const float* __restrict__ bias, OT* __restrict__ out, int lda, int ldo) {
  __shared__ __attribute__((aligned(16))) _Float16 As[128 * 40];
  __shared__ __attribute__((aligned(16))) _Float16 Bs[32 * 72];
  const int tid = threadIdx.x;
  const int lane = tid & 31;
  const int w = tid >> 5;
  const int n0 = blockIdx.x * 64;
  const long m0 = (long)blockIdx.y * 128;

  v8f acc[4] = {};

  for (int k0 = 0; k0 < 512; k0 += 32) {
    if constexpr (sizeof(AT) == 2) {
      // A already f16: DMA 128x32 halves straight into LDS (8-half chunks)
#pragma unroll
      for (int i = 0; i < 2; ++i) {
        int e = i * 256 + tid;          // 0..511 chunks
        int r = e >> 2, c = e & 3;
        async_copy_b128(lds_off(As + r * 40 + c * 8),
                        A + (m0 + r) * lda + k0 + c * 8);
      }
    } else {
#pragma unroll
      for (int i = 0; i < 16; ++i) {    // A: 128x32 f32 -> f16, coalesced
        int idx = i * 256 + tid;
        int r = idx >> 5, c = idx & 31;
        As[r * 40 + c] = (_Float16)(float)A[(m0 + r) * lda + k0 + c];
      }
    }
#pragma unroll
    for (int i = 0; i < 8; ++i) {       // B[k][n] = W[n0+n][k0+k]
      int idx = i * 256 + tid;
      int k = idx & 31, n = idx >> 5;
      Bs[k * 72 + n] = (_Float16)W[(long)(n0 + n) * 512 + k0 + k];
    }
    if constexpr (sizeof(AT) == 2) wait_asynccnt0();
    __syncthreads();
    v16h af = frag_a(As, 16 * w, 40, 0, lane);
#pragma unroll
    for (int t = 0; t < 4; ++t) {
      v16h bf = frag_b(Bs, 0, t * 16, 72, lane);
      acc[t] = wmma16x16x32(af, bf, acc[t]);
    }
    __syncthreads();
  }

  const int lo = lane & 15, hi = lane >> 4;
#pragma unroll
  for (int t = 0; t < 4; ++t)
#pragma unroll
    for (int r = 0; r < 8; ++r) {
      long m = m0 + 16 * w + r + 8 * hi;
      int n = n0 + t * 16 + lo;
      out[m * ldo + n] = (OT)(acc[t][r] + bias[n]);
    }
}

// ---------------------------------------------------------------------------
// Fused cosine attention for one (window b, head h).  q/k/v tiles arrive via
// async DMA into LDS; q/k are L2-normalized in place; S=qn.kn^T and O=P@V run
// on v_wmma_f32_16x16x32_f16; softmax reduces with shfl_xor over the 16-lane
// halves (wave32 C-fragment layout: row m = r + 8*hi, col n = 16*t + lo).
// ---------------------------------------------------------------------------
__global__ __launch_bounds__(128) void attn_kernel(
    const _Float16* __restrict__ qkv, const float* __restrict__ mask,
    const float* __restrict__ t_scale, const float* __restrict__ bias_ws,
    _Float16* __restrict__ attn_out) {
  const int h = blockIdx.x;
  const int b = blockIdx.y;
  const int wi = b % NWIN;
  const int tid = threadIdx.x;
  const int lane = tid & 31;
  const int w = tid >> 5;

  __shared__ __attribute__((aligned(16))) _Float16 qs[64 * 40];
  __shared__ __attribute__((aligned(16))) _Float16 ks[64 * 40];
  __shared__ __attribute__((aligned(16))) _Float16 vs[64 * 40];
  __shared__ __attribute__((aligned(16))) _Float16 ps[64 * 72];

  const long base = ((long)b * NTOK) * QKV_LD + h * HD;
  const _Float16* gbase = qkv + base;

  // Async DMA q,k,v tiles (each 64 rows x 32 halves = 4 chunks/row) into LDS.
  // 768 chunks over 128 threads = 6 per thread.
#pragma unroll
  for (int i = 0; i < 6; ++i) {
    int e = i * 128 + tid;
    int t = e >> 8;               // 0=q, 1=k, 2=v
    int rem = e & 255;
    int row = rem >> 2, c = rem & 3;
    _Float16* larr = (t == 0) ? qs : ((t == 1) ? ks : vs);
    async_copy_b128(lds_off(larr + row * 40 + c * 8),
                    gbase + (long)row * QKV_LD + t * DIM + c * 8);
  }
  wait_asynccnt0();
  __syncthreads();

  {  // threads 0-63: q rows; 64-127: k rows.  L2-normalize in LDS (eps 1e-12).
    int tok = tid & 63;
    _Float16* row = ((tid >> 6) ? ks : qs) + tok * 40;
    float s = 0.f;
#pragma unroll
    for (int d = 0; d < 32; ++d) {
      float v = (float)row[d];
      s += v * v;
    }
    float invn = 1.0f / fmaxf(sqrtf(s), 1e-12f);
#pragma unroll
    for (int d = 0; d < 32; ++d)
      row[d] = (_Float16)((float)row[d] * invn);
  }
  __syncthreads();

  const float scaleh = __expf(fminf(t_scale[h], 4.6051702f));  // log(1/0.01)

  // S = qn @ kn^T : wave w owns rows 16w..16w+15, 4 col tiles, K=32 -> 1 WMMA
  v8f acc[4] = {};
  {
    v16h af = frag_a(qs, 16 * w, 40, 0, lane);
#pragma unroll
    for (int t = 0; t < 4; ++t) {
      v16h bf = frag_bt(ks, 0, t * 16, 40, lane);
      acc[t] = wmma16x16x32(af, bf, acc[t]);
    }
  }

  const int lo = lane & 15, hi = lane >> 4;
  const float* biash = bias_ws + h * 4096;
  const float* maskw = mask + wi * 4096;
#pragma unroll
  for (int t = 0; t < 4; ++t)
#pragma unroll
    for (int r = 0; r < 8; ++r) {
      int m = 16 * w + r + 8 * hi;
      int n = t * 16 + lo;
      acc[t][r] = acc[t][r] * scaleh + biash[m * 64 + n] + maskw[m * 64 + n];
    }

  // softmax: row m lives in lanes with matching hi; reduce 4 in-lane values
  // then xor-shuffle across the 16-lane half (wave32)
#pragma unroll
  for (int r = 0; r < 8; ++r) {
    float mx = fmaxf(fmaxf(acc[0][r], acc[1][r]), fmaxf(acc[2][r], acc[3][r]));
#pragma unroll
    for (int s2 = 1; s2 < 16; s2 <<= 1) mx = fmaxf(mx, __shfl_xor(mx, s2, 32));
    float sum = 0.f;
#pragma unroll
    for (int t = 0; t < 4; ++t) {
      acc[t][r] = __expf(acc[t][r] - mx);
      sum += acc[t][r];
    }
#pragma unroll
    for (int s2 = 1; s2 < 16; s2 <<= 1) sum += __shfl_xor(sum, s2, 32);
    float inv = 1.0f / sum;
#pragma unroll
    for (int t = 0; t < 4; ++t) acc[t][r] *= inv;
  }

  // P -> LDS as f16 (A operand layout for next WMMA)
#pragma unroll
  for (int t = 0; t < 4; ++t)
#pragma unroll
    for (int r = 0; r < 8; ++r)
      ps[(16 * w + r + 8 * hi) * 72 + t * 16 + lo] = (_Float16)acc[t][r];
  __syncthreads();

  // O = P @ V : K=64 (2 steps), 2 col tiles of 16 dims
  v8f oacc[2] = {};
#pragma unroll
  for (int k0 = 0; k0 < 64; k0 += 32) {
    v16h pa = frag_a(ps, 16 * w, 72, k0, lane);
#pragma unroll
    for (int t = 0; t < 2; ++t) {
      v16h vb = frag_b(vs, k0, t * 16, 40, lane);
      oacc[t] = wmma16x16x32(pa, vb, oacc[t]);
    }
  }
#pragma unroll
  for (int t = 0; t < 2; ++t)
#pragma unroll
    for (int r = 0; r < 8; ++r) {
      int m = 16 * w + r + 8 * hi;
      int d = t * 16 + lo;
      attn_out[((long)b * NTOK + m) * DIM + h * HD + d] = (_Float16)oacc[t][r];
    }
}

// ---------------------------------------------------------------------------
extern "C" void kernel_launch(void* const* d_in, const int* in_sizes, int n_in,
                              void* d_out, int out_size, void* d_ws,
                              size_t ws_size, hipStream_t stream) {
  const float* x       = (const float*)d_in[0];
  const float* mask    = (const float*)d_in[1];
  const float* qkv_w   = (const float*)d_in[2];
  const float* qkv_b   = (const float*)d_in[3];
  const float* proj_w  = (const float*)d_in[4];
  const float* proj_b  = (const float*)d_in[5];
  const float* t_scale = (const float*)d_in[6];
  const float* crpb_w1 = (const float*)d_in[7];
  const float* crpb_b1 = (const float*)d_in[8];
  const float* crpb_w2 = (const float*)d_in[9];
  const float* coords  = (const float*)d_in[10];
  const int*   rel_idx = (const int*)d_in[11];

  char* ws = (char*)d_ws;
  _Float16* qkv = (_Float16*)ws;                         // 100352x1536 f16
  size_t off = (size_t)100352 * 1536 * sizeof(_Float16);
  _Float16* attn = (_Float16*)(ws + off);                // 100352x512 f16
  off += (size_t)100352 * 512 * sizeof(_Float16);
  float* bias_ws = (float*)(ws + off);                   // 16x64x64 f32

  bias_kernel<<<dim3(16), dim3(256), 0, stream>>>(coords, crpb_w1, crpb_b1,
                                                  crpb_w2, rel_idx, bias_ws);
  // QKV: M=100352 (784 blocks of 128), N=1536 (24 blocks of 64)
  gemm_bias_kernel<float, _Float16><<<dim3(24, 784), dim3(256), 0, stream>>>(
      x, qkv_w, qkv_b, qkv, 512, 1536);
  // attention: one block per (head, window)
  attn_kernel<<<dim3(16, 1568), dim3(128), 0, stream>>>(qkv, mask, t_scale,
                                                        bias_ws, attn);
  // proj: N=512 (8 blocks of 64)
  gemm_bias_kernel<_Float16, float><<<dim3(8, 784), dim3(256), 0, stream>>>(
      attn, proj_w, proj_b, (float*)d_out, 512, 512);
}
